// _Decoder_27212912787534
// MI455X (gfx1250) — compile-verified, matrix operand order checked by reading, not run on hardware
//
#include <hip/hip_runtime.h>
#include <hip/hip_bf16.h>
#include <math.h>

// CDNA5 wave32 WMMA types
typedef __attribute__((ext_vector_type(16))) __bf16 v16bf;
typedef __attribute__((ext_vector_type(8)))  float  v8f;

#define NB   64                 // persistent blocks (one hidden-tile each for LSTM phases)
#define NT   128                // 4 waves per block
#define NW   (NB * (NT / 32))   // 256 waves total
#define NTH  (NB * NT)          // 8192 threads total

// ---------------- problem dims ----------------
// B=32  T_in=512  T_out=512  n_mel=80  n_enc=512  n_att=128  n_filt=32  K=31
// n_arnn=n_drnn=1024  n_pre=256
// aRNN concat K = 256+512+1024 = 1792 ; dRNN concat K = 1024+512+1024 = 2560

struct Params {
  // inputs
  const float* memory;       // (32,512,512)
  const float* dec_inputs;   // (32,80,512)
  const int*   mem_len;      // (32)
  const float* prenet_w1;    // (256,80)
  const float* prenet_w2;    // (256,256)
  const float* a_wih;        // (4096,768)
  const float* a_whh;        // (4096,1024)
  const float* a_bih;        // (4096)
  const float* a_bhh;        // (4096)
  const float* q_w;          // (128,1024)
  const float* m_w;          // (128,512)
  const float* v_w;          // (1,128)
  const float* conv_w;       // (32,2,31)
  const float* ldw;          // (128,32)
  const float* d_wih;        // (4096,1536)
  const float* d_whh;        // (4096,1024)
  const float* d_bih;        // (4096)
  const float* d_bhh;        // (4096)
  const float* proj_w;       // (80,1536)
  const float* proj_b;       // (80)
  const float* gate_w;       // (1,1536)
  const float* gate_b;       // (1)
  // workspace
  unsigned* bar;             // grid barrier {count, generation}
  __bf16 *Wa;                // (4096,1792) packed [w_ih|w_hh] bf16
  __bf16 *Wd;                // (4096,2560) packed [w_ih|w_hh] bf16
  __bf16 *Wq;                // (128,1024)
  __bf16 *Wm;                // (128,512)
  __bf16 *Wp1;               // (256,96)  K padded 80->96
  __bf16 *Wp2;               // (256,256)
  __bf16 *ldwB;              // (128,32)  loc_dense bf16
  __bf16 *memB;              // (16384,512)  bf16 memory
  __bf16 *x0;                // (16384,96)   teacher-forced prenet input (K padded)
  __bf16 *h1;                // (16384,256)  prenet hidden
  __bf16 *preB;              // (16384,256)  prenet output, row = t*32+b
  __bf16 *xa;                // (32,1792)  [pre_t | actx | ah] staged per step
  __bf16 *xd;                // (32,2560)  [ah | actx | dh]   staged per step
  __bf16 *loccB;             // (16384,32) location conv features (bf16, WMMA A)
  float *biasA, *biasD;      // (4096) combined biases
  float *pm;                 // (16384,128) processed_memory fp32
  float *ah, *ac, *dh, *dc;  // (32,1024) states
  float *aw, *awc, *actx;    // (32,512)x3
  float *pq;                 // (32,128)
  float *eng;                // (16384) attention energies
  // outputs
  float *o_mel;              // (32,80,512)
  float *o_gate;             // (32,512)
  float *o_align;            // (32,512,512)
};

// ---------------- WMMA helpers (A/B: contiguous K per lane half, not ISA table order; C/D per CDNA5 ISA 7.12.2) ----------------

// A fragment 16x32 bf16 (also B fragment 32x16: lane = column, same byte pattern):
// lane L<16 : row/col L, K = k0..k0+15 contiguous ; lane L>=16 : K = k0+16..k0+31
__device__ __forceinline__ v16bf ld_frag(const __bf16* base, int row0, int ldk, int k0) {
  const int lane = threadIdx.x & 31;
  const int r    = lane & 15;
  const int kh   = lane >> 4;
  return *reinterpret_cast<const v16bf*>(base + (size_t)(row0 + r) * ldk + k0 + kh * 16);
}

__device__ __forceinline__ v8f wmma_bf16(v16bf a, v16bf b, v8f c) {
  return __builtin_amdgcn_wmma_f32_16x16x32_bf16(false, a, false, b, (short)0, c, false, false);
}

// C/D layout: VGPR r -> M = r + 8*(lane>>4), N = lane&15
__device__ __forceinline__ void st_tile_f32(float* base, int m0, int n0, int ldn, v8f c) {
  const int lane = threadIdx.x & 31;
  const int n = lane & 15, mh = lane >> 4;
#pragma unroll
  for (int r = 0; r < 8; ++r)
    base[(size_t)(m0 + r + mh * 8) * ldn + n0 + n] = c[r];
}

__device__ __forceinline__ void gsync(unsigned* bar) {
  __syncthreads();
  if (threadIdx.x == 0) {
    unsigned* cnt = bar;
    unsigned* gen = bar + 1;
    unsigned g = __hip_atomic_load(gen, __ATOMIC_ACQUIRE, __HIP_MEMORY_SCOPE_AGENT);
    if (__hip_atomic_fetch_add(cnt, 1u, __ATOMIC_ACQ_REL, __HIP_MEMORY_SCOPE_AGENT) == NB - 1u) {
      __hip_atomic_store(cnt, 0u, __ATOMIC_RELAXED, __HIP_MEMORY_SCOPE_AGENT);
      __hip_atomic_store(gen, g + 1u, __ATOMIC_RELEASE, __HIP_MEMORY_SCOPE_AGENT);
    } else {
      while (__hip_atomic_load(gen, __ATOMIC_ACQUIRE, __HIP_MEMORY_SCOPE_AGENT) == g)
        __builtin_amdgcn_s_sleep(1);
    }
  }
  __syncthreads();
}

__device__ __forceinline__ float sigmf(float x) { return 1.f / (1.f + __expf(-x)); }

// Tiled GEMM, bf16 output: out = act(A(MxK) * W^T), W stored (N,K) row-major
template <bool RELU>
__device__ void gemm_phase_bf16(const __bf16* A, const __bf16* W, __bf16* out,
                                int M, int N, int K, int gw) {
  const int ntn = N >> 4;
  const int ntasks = (M >> 4) * ntn;
  const int lane = threadIdx.x & 31;
  const int n = lane & 15, mh = lane >> 4;
  for (int task = gw; task < ntasks; task += NW) {
    const int m0 = (task / ntn) << 4, n0 = (task % ntn) << 4;
    v8f acc = {};
    for (int k = 0; k < K; k += 32) {
      if (k + 32 < K) __builtin_prefetch(W + (size_t)(n0 + n) * K + k + 32, 0, 1);
      acc = wmma_bf16(ld_frag(A, m0, K, k), ld_frag(W, n0, K, k), acc);
    }
#pragma unroll
    for (int r = 0; r < 8; ++r) {
      float v = acc[r];
      if (RELU) v = fmaxf(v, 0.f);
      out[(size_t)(m0 + r + mh * 8) * N + n0 + n] = (__bf16)v;
    }
  }
}

// Tiled GEMM, fp32 output
__device__ void gemm_phase_f32(const __bf16* A, const __bf16* W, float* out,
                               int M, int N, int K, int gw) {
  const int ntn = N >> 4;
  const int ntasks = (M >> 4) * ntn;
  for (int task = gw; task < ntasks; task += NW) {
    const int m0 = (task / ntn) << 4, n0 = (task % ntn) << 4;
    v8f acc = {};
    for (int k = 0; k < K; k += 32)
      acc = wmma_bf16(ld_frag(A, m0, K, k), ld_frag(W, n0, K, k), acc);
    st_tile_f32(out, m0, n0, N, acc);
  }
}

// Fused LSTM gate GEMM + pointwise: block owns hidden tile h0=blockIdx*16,
// wave w computes gate w (i,f,g,o) for both M-tiles; gates exchanged via LDS.
template <bool STAGE>
__device__ void lstm_phase(const __bf16* X, const __bf16* W, const float* bias,
                           float* cst, float* hst, int Ktot, float* smem,
                           __bf16* hstage, int hstride) {
  const int tid  = threadIdx.x;
  const int wv   = tid >> 5;
  const int lane = tid & 31;
  const int h0   = blockIdx.x * 16;
  const int c0   = wv * 1024 + h0;            // gate column base
  v8f a0 = {}, a1 = {};
  for (int k = 0; k < Ktot; k += 32) {
    if (k + 32 < Ktot) __builtin_prefetch(W + (size_t)(c0 + (lane & 15)) * Ktot + k + 32, 0, 1);
    v16bf bf = ld_frag(W, c0, Ktot, k);
    a0 = wmma_bf16(ld_frag(X, 0,  Ktot, k), bf, a0);
    a1 = wmma_bf16(ld_frag(X, 16, Ktot, k), bf, a1);
  }
  float* g = smem + wv * 512;                 // [32 rows][16 cols]
  const int n = lane & 15, mh = lane >> 4;
#pragma unroll
  for (int r = 0; r < 8; ++r) {
    g[(r + mh * 8) * 16 + n]        = a0[r];
    g[(16 + r + mh * 8) * 16 + n]   = a1[r];
  }
  __syncthreads();
  for (int i = tid; i < 512; i += NT) {       // i = m*16 + nn
    const int m = i >> 4, nn = i & 15;
    const int h = h0 + nn;
    const float zi = smem[0 * 512 + i] + bias[h];
    const float zf = smem[1 * 512 + i] + bias[1024 + h];
    const float zg = smem[2 * 512 + i] + bias[2048 + h];
    const float zo = smem[3 * 512 + i] + bias[3072 + h];
    const float cn = sigmf(zf) * cst[m * 1024 + h] + sigmf(zi) * tanhf(zg);
    const float hn = sigmf(zo) * tanhf(cn);
    cst[m * 1024 + h] = cn;
    hst[m * 1024 + h] = hn;
    if (STAGE) hstage[(size_t)m * hstride + h] = (__bf16)hn;  // bf16 stage for next GEMM
  }
  __syncthreads();
}

__device__ void emit_outputs(const Params& p, int t, int gt) {
  for (int i = gt; i < 32 * 80; i += NTH) {
    const int b = i / 80, m = i % 80;
    const float* wr = p.proj_w + m * 1536;
    const float* hb = p.dh + b * 1024;
    const float* cb = p.actx + b * 512;
    float s = p.proj_b[m];
    for (int k = 0; k < 1024; ++k) s += hb[k] * wr[k];
    for (int k = 0; k < 512;  ++k) s += cb[k] * wr[1024 + k];
    p.o_mel[((size_t)b * 80 + m) * 512 + t] = s;
  }
  if (gt < 32) {
    const int b = gt;
    const float* hb = p.dh + b * 1024;
    const float* cb = p.actx + b * 512;
    float s = p.gate_b[0];
    for (int k = 0; k < 1024; ++k) s += hb[k] * p.gate_w[k];
    for (int k = 0; k < 512;  ++k) s += cb[k] * p.gate_w[1024 + k];
    p.o_gate[b * 512 + t] = s;
  }
}

__global__ __launch_bounds__(NT, 1) void tacotron_decoder(Params p) {
  // 0..8191   : 4 wave-private 16x128 fp32 z-tiles (P5) / LSTM gate exchange (P2,P7)
  //             / conv weights (P4) / softmax buffers (P6)
  // 8192..8319: v_w staging for P5
  __shared__ float smem[8448];
  const int tid  = threadIdx.x;
  const int gt   = blockIdx.x * NT + tid;
  const int wv   = tid >> 5;
  const int gw   = blockIdx.x * 4 + wv;

  // ---------------- W0: pack weights to bf16, stage inputs, zero state ----------------
  for (long i = gt; i < 4096L * 1792; i += NTH) {
    const long nn = i / 1792, k = i % 1792;
    p.Wa[i] = (__bf16)(k < 768 ? p.a_wih[nn * 768 + k] : p.a_whh[nn * 1024 + (k - 768)]);
  }
  for (long i = gt; i < 4096L * 2560; i += NTH) {
    const long nn = i / 2560, k = i % 2560;
    p.Wd[i] = (__bf16)(k < 1536 ? p.d_wih[nn * 1536 + k] : p.d_whh[nn * 1024 + (k - 1536)]);
  }
  for (int i = gt; i < 128 * 1024; i += NTH) p.Wq[i] = (__bf16)p.q_w[i];
  for (int i = gt; i < 128 * 512;  i += NTH) p.Wm[i] = (__bf16)p.m_w[i];
  for (int i = gt; i < 256 * 96;   i += NTH) {
    const int nn = i / 96, k = i % 96;
    p.Wp1[i] = (__bf16)(k < 80 ? p.prenet_w1[nn * 80 + k] : 0.f);
  }
  for (int i = gt; i < 256 * 256;  i += NTH) p.Wp2[i] = (__bf16)p.prenet_w2[i];
  for (int i = gt; i < 128 * 32;   i += NTH) p.ldwB[i] = (__bf16)p.ldw[i];
  for (int i = gt; i < 4096; i += NTH) {
    p.biasA[i] = p.a_bih[i] + p.a_bhh[i];
    p.biasD[i] = p.d_bih[i] + p.d_bhh[i];
  }
  for (long i = gt; i < 32L * 512 * 512; i += NTH) p.memB[i] = (__bf16)p.memory[i];
  for (long i = gt; i < 16384L * 96; i += NTH) {
    const long row = i / 96; const int c = (int)(i % 96);
    const int t = (int)(row >> 5), b = (int)(row & 31);
    float v = 0.f;
    if (c < 80 && t > 0) v = p.dec_inputs[((size_t)b * 80 + c) * 512 + (t - 1)];
    p.x0[i] = (__bf16)v;
  }
  for (int i = gt; i < 32 * 1024; i += NTH) { p.ah[i] = 0; p.ac[i] = 0; p.dh[i] = 0; p.dc[i] = 0; }
  for (int i = gt; i < 32 * 512;  i += NTH) { p.aw[i] = 0; p.awc[i] = 0; p.actx[i] = 0; }
  for (int i = gt; i < 32 * 1792; i += NTH) p.xa[i] = (__bf16)0.f;
  for (int i = gt; i < 32 * 2560; i += NTH) p.xd[i] = (__bf16)0.f;
  gsync(p.bar);

  // ---------------- prenet + processed_memory (WMMA) ----------------
  gemm_phase_bf16<true>(p.x0,  p.Wp1, p.h1,   16384, 256, 96,  gw);  gsync(p.bar);
  gemm_phase_bf16<true>(p.h1,  p.Wp2, p.preB, 16384, 256, 256, gw);  gsync(p.bar);
  gemm_phase_f32(p.memB, p.Wm, p.pm, 16384, 128, 512, gw);           gsync(p.bar);

  // ---------------- sequential decode ----------------
  for (int t = 0; t < 512; ++t) {
    // P1: stage xa=[pre_t|actx|ah], xd dh-region; emit previous step's mel/gate
    for (int i = gt; i < 32 * 1792; i += NTH) {
      const int b = i / 1792, c = i % 1792;
      __bf16 v;
      if (c < 256)      v = p.preB[((size_t)t * 32 + b) * 256 + c];
      else if (c < 768) v = (__bf16)p.actx[b * 512 + (c - 256)];
      else              v = (__bf16)p.ah[b * 1024 + (c - 768)];
      p.xa[i] = v;
    }
    for (int i = gt; i < 32 * 1024; i += NTH) {
      const int b = i >> 10, c = i & 1023;
      p.xd[(size_t)b * 2560 + 1536 + c] = (__bf16)p.dh[i];
    }
    if (t > 0) emit_outputs(p, t - 1, gt);
    gsync(p.bar);

    // P2: attention LSTM (writes ah/ac, stages ah->xd[0:1024) in bf16)
    lstm_phase<true>(p.xa, p.Wa, p.biasA, p.ac, p.ah, 1792, smem, p.xd, 2560);
    gsync(p.bar);

    // P4: query projection (WMMA, blocks 0-1) + location conv -> bf16 (blocks 2-63)
    if (blockIdx.x < 2) {
      const int n0 = (blockIdx.x * 4 + wv) * 16;
      v8f a0 = {}, a1 = {};
      for (int k = 0; k < 1024; k += 32) {
        v16bf bf = ld_frag(p.Wq, n0, 1024, k);
        a0 = wmma_bf16(ld_frag(p.xd, 0,  2560, k), bf, a0);
        a1 = wmma_bf16(ld_frag(p.xd, 16, 2560, k), bf, a1);
      }
      st_tile_f32(p.pq, 0,  n0, 128, a0);
      st_tile_f32(p.pq, 16, n0, 128, a1);
    } else {
      for (int i = tid; i < 1984; i += NT) smem[i] = p.conv_w[i];
      __syncthreads();
      for (int i = (blockIdx.x - 2) * NT + tid; i < 16384; i += 62 * NT) {
        const int b = i >> 9, tt = i & 511;
        float acc[32];
#pragma unroll
        for (int f = 0; f < 32; ++f) acc[f] = 0.f;
        for (int k = 0; k < 31; ++k) {
          const int ti = tt + k - 15;
          if (ti < 0 || ti >= 512) continue;
          const float a0 = p.aw[b * 512 + ti], a1 = p.awc[b * 512 + ti];
#pragma unroll
          for (int f = 0; f < 32; ++f)
            acc[f] += smem[f * 62 + k] * a0 + smem[f * 62 + 31 + k] * a1;
        }
#pragma unroll
        for (int f = 0; f < 32; ++f) p.loccB[(size_t)i * 32 + f] = (__bf16)acc[f];
      }
    }
    gsync(p.bar);

    // P5: z = loc @ ldw^T + pm + pq fully on the matrix pipe (C-operand preloaded),
    //     z-tile parked in wave-private LDS, then e[i] = sum_a v[a]*tanh(z[i,a])
    if (tid < 128) smem[8192 + tid] = p.v_w[tid];
    __syncthreads();
    {
      const int lane = tid & 31;
      const int n = lane & 15, mh = lane >> 4;
      float* zbuf = smem + wv * 2048;                 // 16 x 128 fp32
      for (int task = gw; task < 1024; task += NW) {  // 1024 row-tiles, 4 per wave
        const int m0 = task << 4;
        const int b  = m0 >> 9;                       // tile never crosses batch rows
        const v16bf afrag = ld_frag(p.loccB, m0, 32, 0);   // K=32: single fragment
#pragma unroll
        for (int j = 0; j < 8; ++j) {
          const int n0 = j << 4;
          v8f c;
#pragma unroll
          for (int r = 0; r < 8; ++r)
            c[r] = p.pm[(size_t)(m0 + r + mh * 8) * 128 + n0 + n] + p.pq[b * 128 + n0 + n];
          c = wmma_bf16(afrag, ld_frag(p.ldwB, n0, 32, 0), c);
#pragma unroll
          for (int r = 0; r < 8; ++r)
            zbuf[(r + mh * 8) * 128 + n0 + n] = c[r];
        }
        if (lane < 16) {                              // lanes 0..15: one row each
          const int gi = m0 + lane;
          const int tt = gi & 511;
          float s;
          if (tt >= p.mem_len[b]) s = -1e30f;
          else {
            s = 0.f;
            const float* zr = zbuf + lane * 128;
            for (int a = 0; a < 128; ++a) s += smem[8192 + a] * tanhf(zr[a]);
          }
          p.eng[gi] = s;
        }
      }
    }
    gsync(p.bar);

    // P6: softmax over T_in + context + cumulate + alignment output (blocks 0-31, one per b)
    if (blockIdx.x < 32) {
      const int b = blockIdx.x;
      float* se  = smem;          // 512
      float* red = smem + 512;    // 128
      for (int i = tid; i < 512; i += NT) se[i] = p.eng[b * 512 + i];
      __syncthreads();
      float mx = -1e30f;
      for (int i = tid; i < 512; i += NT) mx = fmaxf(mx, se[i]);
      red[tid] = mx; __syncthreads();
      for (int s2 = 64; s2 > 0; s2 >>= 1) { if (tid < s2) red[tid] = fmaxf(red[tid], red[tid + s2]); __syncthreads(); }
      mx = red[0]; __syncthreads();
      float sm = 0.f;
      for (int i = tid; i < 512; i += NT) { const float ex = __expf(se[i] - mx); se[i] = ex; sm += ex; }
      red[tid] = sm; __syncthreads();
      for (int s2 = 64; s2 > 0; s2 >>= 1) { if (tid < s2) red[tid] += red[tid + s2]; __syncthreads(); }
      const float inv = 1.f / red[0]; __syncthreads();
      for (int i = tid; i < 512; i += NT) {
        const float a = se[i] * inv;
        se[i] = a;
        p.aw[b * 512 + i]  = a;
        p.awc[b * 512 + i] += a;
        p.o_align[((size_t)b * 512 + t) * 512 + i] = a;
      }
      __syncthreads();
      for (int c = tid; c < 512; c += NT) {
        const float* mb = p.memory + (size_t)b * 512 * 512 + c;
        float acc = 0.f;
        for (int ti = 0; ti < 512; ++ti) acc += se[ti] * mb[(size_t)ti * 512];
        p.actx[b * 512 + c] = acc;
        p.xd[(size_t)b * 2560 + 1024 + c] = (__bf16)acc;   // stage for decoder LSTM
      }
    }
    gsync(p.bar);

    // P7: decoder LSTM (writes dh/dc)
    lstm_phase<false>(p.xd, p.Wd, p.biasD, p.dc, p.dh, 2560, smem, nullptr, 0);
    gsync(p.bar);
  }
  emit_outputs(p, 511, gt);
}

extern "C" void kernel_launch(void* const* d_in, const int* in_sizes, int n_in,
                              void* d_out, int out_size, void* d_ws, size_t ws_size,
                              hipStream_t stream) {
  Params p;
  p.memory     = (const float*)d_in[0];
  p.dec_inputs = (const float*)d_in[1];
  p.mem_len    = (const int*)  d_in[2];
  p.prenet_w1  = (const float*)d_in[3];
  p.prenet_w2  = (const float*)d_in[4];
  p.a_wih      = (const float*)d_in[5];
  p.a_whh      = (const float*)d_in[6];
  p.a_bih      = (const float*)d_in[7];
  p.a_bhh      = (const float*)d_in[8];
  p.q_w        = (const float*)d_in[9];
  p.m_w        = (const float*)d_in[10];
  p.v_w        = (const float*)d_in[11];
  p.conv_w     = (const float*)d_in[12];
  p.ldw        = (const float*)d_in[13];
  p.d_wih      = (const float*)d_in[14];
  p.d_whh      = (const float*)d_in[15];
  p.d_bih      = (const float*)d_in[16];
  p.d_bhh      = (const float*)d_in[17];
  p.proj_w     = (const float*)d_in[18];
  p.proj_b     = (const float*)d_in[19];
  p.gate_w     = (const float*)d_in[20];
  p.gate_b     = (const float*)d_in[21];

  char* ws = (char*)d_ws;
  size_t off = 0;
  auto carve = [&](size_t bytes) -> void* {
    void* ptr = ws + off;
    off = (off + bytes + 255) & ~(size_t)255;
    return ptr;
  };
  p.bar   = (unsigned*)carve(256);
  p.Wa    = (__bf16*)carve(4096L * 1792 * 2);
  p.Wd    = (__bf16*)carve(4096L * 2560 * 2);
  p.Wq    = (__bf16*)carve(128L * 1024 * 2);
  p.Wm    = (__bf16*)carve(128L * 512 * 2);
  p.Wp1   = (__bf16*)carve(256L * 96 * 2);
  p.Wp2   = (__bf16*)carve(256L * 256 * 2);
  p.ldwB  = (__bf16*)carve(128L * 32 * 2);
  p.memB  = (__bf16*)carve(32L * 512 * 512 * 2);
  p.x0    = (__bf16*)carve(16384L * 96 * 2);
  p.h1    = (__bf16*)carve(16384L * 256 * 2);
  p.preB  = (__bf16*)carve(16384L * 256 * 2);
  p.xa    = (__bf16*)carve(32L * 1792 * 2);
  p.xd    = (__bf16*)carve(32L * 2560 * 2);
  p.loccB = (__bf16*)carve(16384L * 32 * 2);
  p.biasA = (float*)carve(4096L * 4);
  p.biasD = (float*)carve(4096L * 4);
  p.pm    = (float*)carve(16384L * 128 * 4);
  p.ah    = (float*)carve(32L * 1024 * 4);
  p.ac    = (float*)carve(32L * 1024 * 4);
  p.dh    = (float*)carve(32L * 1024 * 4);
  p.dc    = (float*)carve(32L * 1024 * 4);
  p.aw    = (float*)carve(32L * 512 * 4);
  p.awc   = (float*)carve(32L * 512 * 4);
  p.actx  = (float*)carve(32L * 512 * 4);
  p.pq    = (float*)carve(32L * 128 * 4);
  p.eng   = (float*)carve(16384L * 4);

  float* out = (float*)d_out;
  p.o_mel   = out;                         // (32,80,512)
  p.o_gate  = out + 32L * 80 * 512;        // (32,512)
  p.o_align = p.o_gate + 32L * 512;        // (32,512,512)

  hipMemsetAsync(p.bar, 0, 256, stream);   // reset grid barrier (graph-capturable memset node)
  tacotron_decoder<<<dim3(NB), dim3(NT), 0, stream>>>(p);
}
